// SelfAttention_22462678958098
// MI455X (gfx1250) — matrix operand order repaired and checked by reading, not compiled, not run on hardware
//
#include <hip/hip_runtime.h>

// ---------------------------------------------------------------------------
// Self-attention forward for gfx1250 (MI455X), bf16 WMMA with f32 accumulate.
// B=2, L=2048, E=1024, H=16, HD=64.
// Attention kernel stages K/V tiles into LDS via the Tensor Data Mover
// (tensor_load_to_lds, TENSORcnt double-buffering).
// ---------------------------------------------------------------------------

typedef __attribute__((ext_vector_type(16))) __bf16 v16bf;
typedef __attribute__((ext_vector_type(8)))  __bf16 bf16x8;
typedef __attribute__((ext_vector_type(8)))  float  v8f;
typedef __attribute__((ext_vector_type(8)))  float  f32x8;
typedef __attribute__((ext_vector_type(4)))  unsigned int u32x4;
typedef __attribute__((ext_vector_type(8)))  int i32x8;
typedef __attribute__((ext_vector_type(4)))  int i32x4;

#define B_  2
#define L_  2048
#define E_  1024
#define H_  16
#define HD_ 64
#define N1_ (3 * E_)   // 3072

static __device__ __forceinline__ v8f wmma_bf16(v16bf a, v16bf b, v8f c) {
    // (neg_a, A, neg_b, B, c_mod, C, reuse_a, reuse_b)
    return __builtin_amdgcn_wmma_f32_16x16x32_bf16(false, a, false, b,
                                                   (short)0, c, false, false);
}

static __device__ __forceinline__ v16bf cat16(bf16x8 lo, bf16x8 hi) {
    return __builtin_shufflevector(lo, hi, 0, 1, 2, 3, 4, 5, 6, 7,
                                           8, 9, 10, 11, 12, 13, 14, 15);
}
static __device__ __forceinline__ v16bf cvt16(f32x8 lo, f32x8 hi) {
    v16bf r;
    #pragma unroll
    for (int i = 0; i < 8; ++i) { r[i] = (__bf16)lo[i]; r[8 + i] = (__bf16)hi[i]; }
    return r;
}

// A-fragment (16x32, 16-bit): per lane, elements 0..15 = K(half*16 + 0..15).
static __device__ __forceinline__ v16bf loadA_f32(const float* row, int half) {
    f32x8 lo = *(const f32x8*)(row + half * 16);
    f32x8 hi = *(const f32x8*)(row + half * 16 + 8);
    return cvt16(lo, hi);
}
static __device__ __forceinline__ v16bf loadA_bf16(const __bf16* row, int half) {
    return *(const v16bf*)(row + half * 16);
}
// B-fragment (32x16, 16-bit): per lane, elements 0..15 = K(half*16 + 0..15).
static __device__ __forceinline__ v16bf loadB_f32(const float* row, int half) {
    f32x8 lo = *(const f32x8*)(row + half * 16);
    f32x8 hi = *(const f32x8*)(row + half * 16 + 8);
    return cvt16(lo, hi);
}
static __device__ __forceinline__ v16bf loadB_bf16(const __bf16* row, int half) {
    return *(const v16bf*)(row + half * 16);
}

// ---------------------------------------------------------------------------
// TDM: issue a 2D tile load (Global -> LDS).  D# group0/group1 per CDNA5 ISA
// 8.3/8.4; groups 2/3 zero (2D tensor).  data_size = 2 bytes (bf16).
// dims/strides in elements.  EXEC is ignored; issue from one wave only.
// 6-arg builtin form: (g0, g1, g2, g3, g4, cpol).
// ---------------------------------------------------------------------------
static __device__ __forceinline__ void tdm_load_2d(
    unsigned lds_addr, const void* gaddr,
    unsigned tile_d0, unsigned tile_d1,
    unsigned long long tensor_d0, unsigned long long stride0)
{
    unsigned long long ga = (unsigned long long)(size_t)gaddr;
    u32x4 g0;
    g0[0] = 1u;                                          // count=1 (valid user D#)
    g0[1] = lds_addr;                                    // LDS byte address
    g0[2] = (unsigned)(ga & 0xFFFFFFFFu);                // global_addr[31:0]
    g0[3] = (unsigned)((ga >> 32) & 0x01FFFFFFu)         // global_addr[56:32]
          | (2u << 30);                                  // type=2 ("image")
    // group1, assembled as 4x 64-bit quads (bit positions per ISA 8.4)
    unsigned long long q0 = (1ull << 16)                 // data_size code 1 = 2B
                          | ((tensor_d0 & 0xFFFFull) << 48);       // dim0[15:0]
    unsigned long long q1 = ((tensor_d0 >> 16) & 0xFFFFull)        // dim0[31:16]
                          | ((unsigned long long)tile_d1 << 16)    // tensor_dim1
                          | ((unsigned long long)(tile_d0 & 0xFFFFu) << 48); // tile_dim0
    unsigned long long q2 = (unsigned long long)(tile_d1 & 0xFFFFu)          // tile_dim1
                          | ((stride0 & 0xFFFFFFFFull) << 32);     // stride0[31:0]
    unsigned long long q3 = ((stride0 >> 32) & 0xFFFFull);         // stride0[47:32]
    i32x8 g1;
    g1[0] = (int)q0; g1[1] = (int)(q0 >> 32);
    g1[2] = (int)q1; g1[3] = (int)(q1 >> 32);
    g1[4] = (int)q2; g1[5] = (int)(q2 >> 32);
    g1[6] = (int)q3; g1[7] = (int)(q3 >> 32);
    i32x4 z4 = {0, 0, 0, 0};
    i32x8 z8 = {0, 0, 0, 0, 0, 0, 0, 0};
    __builtin_amdgcn_tensor_load_to_lds(g0, g1, z4, z4, z8, 0);
}

// ---------------------------------------------------------------------------
// Kernel 1: fused QKV projection.  C[m,n] = sum_e x[m,e] * Wqkv[n,e] + bqkv[n]
// One wave per 16(M) x 64(N) tile (one head-column), K loop in steps of 32.
// Writes Q (pre-scaled by 1/sqrt(HD)) as [BH, L, HD] bf16,
//        K as [BH, L, HD] bf16, and V transposed as [BH, HD, L] bf16.
// ---------------------------------------------------------------------------
__global__ __launch_bounds__(256) void qkv_kernel(
    const float* __restrict__ x, const float* __restrict__ Wqkv,
    const float* __restrict__ bqkv,
    __bf16* __restrict__ Qbf, __bf16* __restrict__ Kbf, __bf16* __restrict__ Vt)
{
    const int wave = threadIdx.x >> 5;
    const int lane = threadIdx.x & 31;
    const int half = lane >> 4, l15 = lane & 15;

    const int NT = N1_ / 64;                    // 48 column supertiles
    int tile = blockIdx.x * 8 + wave;           // 12288 tiles total
    int mt = tile / NT, nt = tile % NT;

    const float* xrow = x    + (size_t)(mt * 16 + l15) * E_;
    const float* wrow = Wqkv + (size_t)(nt * 64 + l15) * E_;

    v8f c0 = {}, c1 = {}, c2 = {}, c3 = {};
    for (int k0 = 0; k0 < E_; k0 += 32) {
        v16bf a = loadA_f32(xrow + k0, half);
        c0 = wmma_bf16(a, loadB_f32(wrow + 0 * 16 * E_ + k0, half), c0);
        c1 = wmma_bf16(a, loadB_f32(wrow + 1 * 16 * E_ + k0, half), c1);
        c2 = wmma_bf16(a, loadB_f32(wrow + 2 * 16 * E_ + k0, half), c2);
        c3 = wmma_bf16(a, loadB_f32(wrow + 3 * 16 * E_ + k0, half), c3);
    }

    int ncol0 = nt * 64;
    int part  = ncol0 >> 10;                    // wave-uniform: 0=Q,1=K,2=V
    int h     = (ncol0 & (E_ - 1)) >> 6;        // wave-uniform head

    #pragma unroll
    for (int s = 0; s < 4; ++s) {
        const v8f& c = s == 0 ? c0 : s == 1 ? c1 : s == 2 ? c2 : c3;
        int d = s * 16 + l15;                   // within-head dim 0..63
        float bias = bqkv[ncol0 + d];
        #pragma unroll
        for (int r = 0; r < 8; ++r) {
            int m  = mt * 16 + r + half * 8;
            int bb = m >> 11, l = m & (L_ - 1);
            size_t bh = (size_t)(bb * H_ + h);
            float v = c[r] + bias;
            if (part == 0)      Qbf[(bh * L_ + l) * HD_ + d] = (__bf16)(v * 0.125f);
            else if (part == 1) Kbf[(bh * L_ + l) * HD_ + d] = (__bf16)v;
            else                Vt [(bh * HD_ + d) * L_ + l] = (__bf16)v;
        }
    }
}

// ---------------------------------------------------------------------------
// Kernel 2: flash attention.  8 waves/block share one (b,h); consecutive
// query tiles.  K (32x64) and V^T (64x32) tiles staged into LDS by the TDM,
// double-buffered on TENSORcnt; waves read fragments from LDS.
// ---------------------------------------------------------------------------
__global__ __launch_bounds__(256) void attn_kernel(
    const __bf16* __restrict__ Qbf, const __bf16* __restrict__ Kbf,
    const __bf16* __restrict__ Vt, __bf16* __restrict__ attn)
{
    __shared__ __align__(32) __bf16 Kl[2][32 * 64];     // 2 x 4 KB key tiles
    __shared__ __align__(32) __bf16 Vl[2][64 * 32];     // 2 x 4 KB V^T tiles
    __shared__ __align__(32) __bf16 Pst[8][16][32];     // per-wave P staging

    const int wave = threadIdx.x >> 5;
    const int lane = threadIdx.x & 31;
    const int half = lane >> 4, l15 = lane & 15;

    int wid  = blockIdx.x * 8 + wave;           // 4096 waves
    int qt   = wid & 127;                       // query tile within (b,h)
    int bh   = wid >> 7;                        // 0..31, block-uniform
    int qbase = qt * 16;

    const __bf16* Qp = Qbf + ((size_t)bh * L_ + qbase) * HD_;
    const __bf16* Kp = Kbf + (size_t)bh * L_ * HD_;
    const __bf16* Vp = Vt  + (size_t)bh * HD_ * L_;

    // Preload Q fragments (16x64 = two K=32 A-fragments), already scaled.
    const __bf16* qrow = Qp + (size_t)l15 * HD_;
    v16bf qa0 = loadA_bf16(qrow, half);
    v16bf qa1 = loadA_bf16(qrow + 32, half);

    v8f o0 = {}, o1 = {}, o2 = {}, o3 = {};
    float mrow[8], lrow[8];
    #pragma unroll
    for (int r = 0; r < 8; ++r) { mrow[r] = -1e30f; lrow[r] = 0.f; }

    const int NKT = L_ / 32;                    // 64 key tiles
    // Prologue: TDM-load tile 0 into buffer 0 (wave 0 only; EXEC ignored).
    if (wave == 0) {
        tdm_load_2d((unsigned)(size_t)&Kl[0][0], Kp, 64, 32, 64, 64);
        tdm_load_2d((unsigned)(size_t)&Vl[0][0], Vp, 32, 64, 32, L_);
    }

    for (int i = 0; i < NKT; ++i) {
        int cur = i & 1;
        int kt  = i * 32;
        if (wave == 0) {
            if (i + 1 < NKT) {
                int ktn = kt + 32;
                tdm_load_2d((unsigned)(size_t)&Kl[cur ^ 1][0],
                            Kp + (size_t)ktn * HD_, 64, 32, 64, 64);
                tdm_load_2d((unsigned)(size_t)&Vl[cur ^ 1][0],
                            Vp + ktn, 32, 64, 32, L_);
                __builtin_amdgcn_s_wait_tensorcnt(2);   // tile i complete
            } else {
                __builtin_amdgcn_s_wait_tensorcnt(0);
            }
        }
        __syncthreads();                        // tile i visible to all waves

        // ---- scores: S[16x32] = Q(16x64) * K^T  (4 WMMAs, K from LDS) ----
        const __bf16* kr0 = &Kl[cur][(size_t)l15 * HD_];
        const __bf16* kr1 = kr0 + 16 * HD_;
        v8f sA = {}, sB = {};
        sA = wmma_bf16(qa0, loadB_bf16(kr0, half), sA);
        sB = wmma_bf16(qa0, loadB_bf16(kr1, half), sB);
        sA = wmma_bf16(qa1, loadB_bf16(kr0 + 32, half), sA);
        sB = wmma_bf16(qa1, loadB_bf16(kr1 + 32, half), sB);

        // ---- online softmax (row = r + half*8, spread over 16 lanes) ----
        float alpha[8];
        #pragma unroll
        for (int r = 0; r < 8; ++r) {
            float mx = fmaxf(sA[r], sB[r]);
            mx = fmaxf(mx, __shfl_xor(mx, 8, 32));
            mx = fmaxf(mx, __shfl_xor(mx, 4, 32));
            mx = fmaxf(mx, __shfl_xor(mx, 2, 32));
            mx = fmaxf(mx, __shfl_xor(mx, 1, 32));
            float mnew = fmaxf(mrow[r], mx);
            alpha[r] = __expf(mrow[r] - mnew);
            float pa = __expf(sA[r] - mnew);
            float pb = __expf(sB[r] - mnew);
            float rs = pa + pb;
            rs += __shfl_xor(rs, 8, 32);
            rs += __shfl_xor(rs, 4, 32);
            rs += __shfl_xor(rs, 2, 32);
            rs += __shfl_xor(rs, 1, 32);
            lrow[r] = lrow[r] * alpha[r] + rs;
            mrow[r] = mnew;
            int row = r + half * 8;
            Pst[wave][row][l15]      = (__bf16)pa;   // keys [kt, kt+16)
            Pst[wave][row][16 + l15] = (__bf16)pb;   // keys [kt+16, kt+32)
            o0[r] *= alpha[r]; o1[r] *= alpha[r];
            o2[r] *= alpha[r]; o3[r] *= alpha[r];
        }

        // ---- reload P in A-fragment layout (same-wave LDS is in-order) ----
        v16bf pfrag = loadA_bf16(&Pst[wave][l15][0], half);

        // ---- O(16x64) += P(16x32) * V(32x64)  (4 WMMAs, V^T from LDS) ----
        o0 = wmma_bf16(pfrag, loadB_bf16(&Vl[cur][( 0 + l15) * 32], half), o0);
        o1 = wmma_bf16(pfrag, loadB_bf16(&Vl[cur][(16 + l15) * 32], half), o1);
        o2 = wmma_bf16(pfrag, loadB_bf16(&Vl[cur][(32 + l15) * 32], half), o2);
        o3 = wmma_bf16(pfrag, loadB_bf16(&Vl[cur][(48 + l15) * 32], half), o3);

        __syncthreads();    // all waves done with this buffer before overwrite
    }

    // ---- normalize + store attn[b, l, h*64 + d] as bf16 ----
    int bb = bh >> 4, h = bh & 15;
    #pragma unroll
    for (int r = 0; r < 8; ++r) {
        int row = qbase + r + half * 8;
        float inv = 1.0f / lrow[r];
        size_t base = ((size_t)(bb * L_ + row)) * E_ + h * HD_;
        attn[base + 0  + l15] = (__bf16)(o0[r] * inv);
        attn[base + 16 + l15] = (__bf16)(o1[r] * inv);
        attn[base + 32 + l15] = (__bf16)(o2[r] * inv);
        attn[base + 48 + l15] = (__bf16)(o3[r] * inv);
    }
}

// ---------------------------------------------------------------------------
// Kernel 3: output projection.  out[m,n] = sum_e attn[m,e]*Wout[n,e] + bout[n]
// ---------------------------------------------------------------------------
__global__ __launch_bounds__(256) void out_kernel(
    const __bf16* __restrict__ attn, const float* __restrict__ Wout,
    const float* __restrict__ bout, float* __restrict__ out)
{
    const int wave = threadIdx.x >> 5;
    const int lane = threadIdx.x & 31;
    const int half = lane >> 4, l15 = lane & 15;

    const int NT = E_ / 64;                     // 16 column supertiles
    int tile = blockIdx.x * 8 + wave;           // 4096 tiles
    int mt = tile / NT, nt = tile % NT;

    const __bf16* arow = attn + (size_t)(mt * 16 + l15) * E_;
    const float*  wrow = Wout + (size_t)(nt * 64 + l15) * E_;

    v8f c0 = {}, c1 = {}, c2 = {}, c3 = {};
    for (int k0 = 0; k0 < E_; k0 += 32) {
        v16bf a = loadA_bf16(arow + k0, half);
        c0 = wmma_bf16(a, loadB_f32(wrow + 0 * 16 * E_ + k0, half), c0);
        c1 = wmma_bf16(a, loadB_f32(wrow + 1 * 16 * E_ + k0, half), c1);
        c2 = wmma_bf16(a, loadB_f32(wrow + 2 * 16 * E_ + k0, half), c2);
        c3 = wmma_bf16(a, loadB_f32(wrow + 3 * 16 * E_ + k0, half), c3);
    }

    #pragma unroll
    for (int s = 0; s < 4; ++s) {
        const v8f& c = s == 0 ? c0 : s == 1 ? c1 : s == 2 ? c2 : c3;
        int ncol = nt * 64 + s * 16 + l15;
        float bias = bout[ncol];
        #pragma unroll
        for (int r = 0; r < 8; ++r) {
            int m = mt * 16 + r + half * 8;
            out[(size_t)m * E_ + ncol] = c[r] + bias;
        }
    }
}

// ---------------------------------------------------------------------------
extern "C" void kernel_launch(void* const* d_in, const int* in_sizes, int n_in,
                              void* d_out, int out_size, void* d_ws, size_t ws_size,
                              hipStream_t stream) {
    (void)in_sizes; (void)n_in; (void)out_size; (void)ws_size;
    const float* x    = (const float*)d_in[0];
    const float* Wqkv = (const float*)d_in[1];
    const float* bqkv = (const float*)d_in[2];
    const float* Wout = (const float*)d_in[3];
    const float* bout = (const float*)d_in[4];
    float* out = (float*)d_out;

    // Workspace layout (bf16): Q | K | V^T | attn  (4M elems each = 32 MB)
    size_t perbuf = (size_t)B_ * H_ * L_ * HD_;
    __bf16* Qbf  = (__bf16*)d_ws;
    __bf16* Kbf  = Qbf + perbuf;
    __bf16* Vt   = Kbf + perbuf;
    __bf16* attn = Vt + perbuf;

    qkv_kernel<<<1536, 256, 0, stream>>>(x, Wqkv, bqkv, Qbf, Kbf, Vt);
    attn_kernel<<<512, 256, 0, stream>>>(Qbf, Kbf, Vt, attn);
    out_kernel<<<512, 256, 0, stream>>>(attn, Wout, bout, out);
}
